// CoreAttention_49203145343015
// MI455X (gfx1250) — matrix operand order repaired and checked by reading, not compiled, not run on hardware
//
#include <hip/hip_runtime.h>

// Causal multi-head attention (flash-attention style) for MI455X / gfx1250.
// Shapes: SQ=SK=2048, B=2, NP=16, HN=128, fp32 in/out, f16 WMMA compute.
// Block = 256 threads = 8 wave32 (2 waves/SIMD on a WGP); each wave owns a
// 16-row Q slab of a 128-row Q tile; K/V staged once per tile for all 8 waves.
// Causal structure exploited dynamically: tiles above a wave's diagonal are
// skipped entirely; tiles below it take a mask-free softmax fast path.

#define SQL 2048
#define SKL 2048
#define NB 2
#define NHEAD 16
#define HND 128
#define ROWSTRIDE (NB * NHEAD * HND) /* 4096 floats between seq positions */
#define QTILE 128
#define NWAVES 8

typedef __attribute__((ext_vector_type(16))) _Float16 v16h;
typedef __attribute__((ext_vector_type(8)))  _Float16 v8h;
typedef __attribute__((ext_vector_type(4)))  _Float16 v4h;
typedef __attribute__((ext_vector_type(8)))  float    v8f;

// 32-byte (16-half) load built from two 16-byte chunks (only 16B alignment needed)
static __device__ __forceinline__ v16h ld16h(const _Float16* p) {
  v8h lo = *(const v8h*)p;
  v8h hi = *(const v8h*)(p + 8);
  return __builtin_shufflevector(lo, hi, 0, 1, 2, 3, 4, 5, 6, 7,
                                 8, 9, 10, 11, 12, 13, 14, 15);
}

// Online-softmax update for one 32-key tile. C-layout: lane=key-col, vgpr=row.
// MASKED=false is the fast path for tiles entirely below the causal diagonal.
template <bool MASKED>
static __device__ __forceinline__ void softmax_update(
    const v8f& s0, const v8f& s1, int kb, int myrow, int ln15, int hi, int lane,
    float* mrow, float* lrow, float* alpha, _Float16 (*sPw)[16]) {
  const int col0 = kb + ln15;
  const int col1 = col0 + 16;
#pragma unroll
  for (int r = 0; r < 8; ++r) {
    float x0, x1;
    if (MASKED) {
      const int rr = myrow + r;
      x0 = (col0 > rr) ? -10000.f : s0[r];
      x1 = (col1 > rr) ? -10000.f : s1[r];
    } else {
      x0 = s0[r];
      x1 = s1[r];
    }
    float t = fmaxf(x0, x1);
    t = fmaxf(t, __shfl_xor(t, 1, 32));
    t = fmaxf(t, __shfl_xor(t, 2, 32));
    t = fmaxf(t, __shfl_xor(t, 4, 32));
    t = fmaxf(t, __shfl_xor(t, 8, 32));
    const float mnew = fmaxf(mrow[r], t);
    const float al = __expf(mrow[r] - mnew);
    const float p0 = __expf(x0 - mnew);
    const float p1 = __expf(x1 - mnew);
    float rs = p0 + p1;
    rs += __shfl_xor(rs, 1, 32);
    rs += __shfl_xor(rs, 2, 32);
    rs += __shfl_xor(rs, 4, 32);
    rs += __shfl_xor(rs, 8, 32);
    lrow[r]  = lrow[r] * al + rs;
    mrow[r]  = mnew;
    alpha[r] = al;
    // scatter into A-layout: a-lane = row + 16*(key>=16); slot = key&15
    const int la = r + 8 * hi;
    sPw[la][lane & 15]      = (_Float16)p0;
    sPw[la + 16][lane & 15] = (_Float16)p1;
  }
}

__launch_bounds__(256, 1)
__global__ void core_attention_fa_kernel(const float* __restrict__ Q,
                                         const float* __restrict__ K,
                                         const float* __restrict__ V,
                                         float* __restrict__ O) {
  // K tile: [key 0..31][h 0..127], row padded to 136 halves (272B) for bank spread
  __shared__ _Float16 sK[32][HND + 8];
  // V tile transposed: [h 0..127][key 0..31], row padded to 40 halves (80B)
  __shared__ _Float16 sV[HND][32 + 8];
  // Per-wave P staging in A-matrix layout: [wave][a-lane 0..31][16 halves]
  __shared__ _Float16 sP[NWAVES][32][16];

  const int tid  = threadIdx.x;
  const int lane = tid & 31;
  const int wave = tid >> 5;
  const int ln15 = lane & 15;
  const int hi   = lane >> 4; // which half of the wave (K/N upper half selector)

  const int  rowbase = blockIdx.x * QTILE;              // 128-row Q tile
  const long headoff = (long)blockIdx.z * (NHEAD * HND) // batch
                     + (long)blockIdx.y * HND;          // head

  const float scale = 0.08838834764831845f; // 1/sqrt(128)

  // ---- Q fragments: A-matrix layout (16x32 f16 per 32-wide K chunk), pre-scaled.
  // lanes 0-15: row M=lane, k in {0..15}; lanes 16-31: row M=lane-16, k in {16..31}
  v16h qa[4];
  {
    const int grow = rowbase + wave * 16 + ln15;
    const float* qp = Q + (long)grow * ROWSTRIDE + headoff;
#pragma unroll
    for (int c = 0; c < 4; ++c) {
      const int s0 = 32 * c + 16 * hi;
      const int s1 = s0 + 8;
      v16h a;
#pragma unroll
      for (int i = 0; i < 8; ++i) {
        a[i]     = (_Float16)(qp[s0 + i] * scale);
        a[8 + i] = (_Float16)(qp[s1 + i] * scale);
      }
      qa[c] = a;
    }
  }

  // Output accumulators: 8 N-tiles of 16 columns, C-layout (lane=N, vgpr=row)
  v8f acc[8];
#pragma unroll
  for (int nt = 0; nt < 8; ++nt) acc[nt] = (v8f){0, 0, 0, 0, 0, 0, 0, 0};
  float mrow[8], lrow[8];
#pragma unroll
  for (int r = 0; r < 8; ++r) { mrow[r] = -1e30f; lrow[r] = 0.f; }

  const int myrow = rowbase + wave * 16 + 8 * hi; // row of acc[*][r] is myrow + r
  const int wmin  = rowbase + wave * 16;          // this wave's lowest row
  const int wmax  = wmin + 15;                    // this wave's highest row
  const int kend  = rowbase + QTILE;              // block-wide causal limit

  // LDS staging: 256 threads move the 32x128 tile; thread -> (key row, 16-col slice)
  const int srow = tid >> 3;          // 0..31
  const int scol = (tid & 7) * 16;    // 0,16,...,112

  for (int kb = 0; kb < kend; kb += 32) {
    // ---- stage K,V tile into LDS (fp32 -> f16; V transposed) ----
    {
      const float4* kp4 = (const float4*)(K + (long)(kb + srow) * ROWSTRIDE + headoff + scol);
      const float4* vp4 = (const float4*)(V + (long)(kb + srow) * ROWSTRIDE + headoff + scol);
#pragma unroll
      for (int i = 0; i < 4; ++i) {
        float4 kq = kp4[i];
        v4h kh;
        kh[0] = (_Float16)kq.x; kh[1] = (_Float16)kq.y;
        kh[2] = (_Float16)kq.z; kh[3] = (_Float16)kq.w;
        *(v4h*)&sK[srow][scol + 4 * i] = kh;
        float4 vq = vp4[i];
        sV[scol + 4 * i + 0][srow] = (_Float16)vq.x;
        sV[scol + 4 * i + 1][srow] = (_Float16)vq.y;
        sV[scol + 4 * i + 2][srow] = (_Float16)vq.z;
        sV[scol + 4 * i + 3][srow] = (_Float16)vq.w;
      }
    }
    __syncthreads();

    // Prefetch next K/V tile toward WGP$/L2 while we compute on this one
    if (kb + 32 < kend) {
      __builtin_prefetch(K + (long)(kb + 32 + srow) * ROWSTRIDE + headoff + scol, 0, 0);
      __builtin_prefetch(V + (long)(kb + 32 + srow) * ROWSTRIDE + headoff + scol, 0, 0);
    }

    // Wave-uniform causal skip: tiles entirely above this wave's diagonal
    // contribute exp(-10000-m) ~ 0; skip all compute (EXEC stays all-ones
    // inside, so WMMA's EXEC restriction is respected on the taken path).
    if (kb <= wmax) {
      // ---- S = (Q*scale) K^T : two 16-key N-tiles, K-dim 128 in 4 chunks ----
      v8f s0 = (v8f){0, 0, 0, 0, 0, 0, 0, 0};
      v8f s1 = (v8f){0, 0, 0, 0, 0, 0, 0, 0};
#pragma unroll
      for (int c = 0; c < 4; ++c) {
        // B fragment: lane holds key-column (N)=ln15(+16), k range = 16*hi..+15
        v16h b0 = ld16h(&sK[ln15][32 * c + 16 * hi]);
        v16h b1 = ld16h(&sK[16 + ln15][32 * c + 16 * hi]);
        s0 = __builtin_amdgcn_wmma_f32_16x16x32_f16(false, qa[c], false, b0,
                                                    (short)0, s0, false, false);
        s1 = __builtin_amdgcn_wmma_f32_16x16x32_f16(false, qa[c], false, b1,
                                                    (short)0, s1, false, false);
      }

      // ---- online softmax: mask-free fast path below the diagonal ----
      float alpha[8];
      if (kb + 31 <= wmin) {
        softmax_update<false>(s0, s1, kb, myrow, ln15, hi, lane,
                              mrow, lrow, alpha, sP[wave]);
      } else {
        softmax_update<true>(s0, s1, kb, myrow, ln15, hi, lane,
                             mrow, lrow, alpha, sP[wave]);
      }

      // ---- O = diag(alpha)*O + P * V ----
      v16h pa = ld16h(&sP[wave][lane][0]); // contiguous A-fragment readback
#pragma unroll
      for (int nt = 0; nt < 8; ++nt) {
        v8f a = acc[nt];
#pragma unroll
        for (int r = 0; r < 8; ++r) a[r] *= alpha[r];
        // B fragment of V: N (=h within tile) = ln15, k (=key) range 16*hi..+15
        v16h bv = ld16h(&sV[16 * nt + ln15][16 * hi]);
        acc[nt] = __builtin_amdgcn_wmma_f32_16x16x32_f16(false, pa, false, bv,
                                                         (short)0, a, false, false);
      }
    }
    __syncthreads();
  }

  // ---- epilogue: normalize by running sum, store fp32 [sq, b, np*hn] ----
#pragma unroll
  for (int r = 0; r < 8; ++r) {
    const float inv = 1.f / lrow[r];
    float* op = O + (long)(myrow + r) * ROWSTRIDE + headoff + ln15;
#pragma unroll
    for (int nt = 0; nt < 8; ++nt) {
      op[16 * nt] = acc[nt][r] * inv;
    }
  }
}

extern "C" void kernel_launch(void* const* d_in, const int* in_sizes, int n_in,
                              void* d_out, int out_size, void* d_ws, size_t ws_size,
                              hipStream_t stream) {
  (void)in_sizes; (void)n_in; (void)d_ws; (void)ws_size; (void)out_size;
  const float* q = (const float*)d_in[0];
  const float* k = (const float*)d_in[1];
  const float* v = (const float*)d_in[2];
  // d_in[3] is the causal bool mask; computed analytically in-kernel.
  float* out = (float*)d_out;
  dim3 grid(SQL / QTILE, NHEAD, NB);
  core_attention_fa_kernel<<<grid, dim3(256), 0, stream>>>(q, k, v, out);
}
